// MultiHeadAttention_85117661872728
// MI455X (gfx1250) — matrix operand order repaired and checked by reading, not compiled, not run on hardware
//
#include <hip/hip_runtime.h>
#include <hip/hip_bf16.h>

// ---- CDNA5 WMMA types -------------------------------------------------------
typedef __attribute__((ext_vector_type(16))) __bf16 v16bf;
typedef __attribute__((ext_vector_type(8)))  float  v8f;

union BF16x16 { v16bf v; unsigned short s[16]; };

__device__ __forceinline__ unsigned short f2bfu(float f) {
    union { float f; unsigned u; } x; x.f = f;
    unsigned r = x.u + 0x7FFFu + ((x.u >> 16) & 1u);   // round-to-nearest-even
    return (unsigned short)(r >> 16);
}

__device__ __forceinline__ float rmax16(float v) {
    v = fmaxf(v, __shfl_xor(v, 1, 32));
    v = fmaxf(v, __shfl_xor(v, 2, 32));
    v = fmaxf(v, __shfl_xor(v, 4, 32));
    v = fmaxf(v, __shfl_xor(v, 8, 32));
    return v;
}
__device__ __forceinline__ float rsum16(float v) {
    v += __shfl_xor(v, 1, 32);
    v += __shfl_xor(v, 2, 32);
    v += __shfl_xor(v, 4, 32);
    v += __shfl_xor(v, 8, 32);
    return v;
}

// ---- gfx1250 async global->LDS copy (ASYNCcnt path, CDNA5 ISA) -----
__device__ __forceinline__ void async_copy_b128(unsigned lds_off, const void* gaddr) {
    asm volatile("global_load_async_to_lds_b128 %0, %1, off"
                 :
                 : "v"(lds_off), "v"((unsigned long long)(uintptr_t)gaddr)
                 : "memory");
}
__device__ __forceinline__ void wait_async0() {
    asm volatile("s_wait_asynccnt 0" ::: "memory");
}
__device__ __forceinline__ unsigned lds_off_of(const void* p) {
    return (unsigned)(uintptr_t)p;   // low 32 bits of generic shared ptr = LDS offset
}

// ---- one-time fp32 -> bf16 conversion (elementwise, vectorized) -------------
__global__ __launch_bounds__(256) void cvt_bf16_kernel(
    const float4* __restrict__ src, uint2* __restrict__ dst, int n4)
{
    int i = blockIdx.x * blockDim.x + threadIdx.x;
    if (i >= n4) return;
    float4 f = src[i];
    uint2 o;
    o.x = ((unsigned)f2bfu(f.y) << 16) | f2bfu(f.x);
    o.y = ((unsigned)f2bfu(f.w) << 16) | f2bfu(f.z);
    dst[i] = o;
}

// ---- one-time fp32 [K,N] -> bf16 [N,K] transpose (LDS tiled) ----------------
__global__ __launch_bounds__(256) void cvt_transpose_kernel(
    const float* __restrict__ W, unsigned short* __restrict__ Wt, int K, int N)
{
    __shared__ unsigned short T[64][65];
    const int k0 = blockIdx.y * 64, n0 = blockIdx.x * 64;
    const int tid = threadIdx.x;
    #pragma unroll
    for (int j = 0; j < 16; ++j) {
        int idx = j * 256 + tid;
        int r = idx >> 6, c = idx & 63;
        T[r][c] = f2bfu(W[(size_t)(k0 + r) * N + n0 + c]);
    }
    __syncthreads();
    #pragma unroll
    for (int j = 0; j < 16; ++j) {
        int idx = j * 256 + tid;
        int r = idx >> 6, c = idx & 63;
        Wt[(size_t)(n0 + r) * K + k0 + c] = T[c][r];
    }
}

// ---- GEMM: Y[M,N] = X[M,K](bf16) * Wt[N,K](bf16)^T, +bias, *outScale --------
// 64x64 block tile, BK=32, 256 thr = 8 waves (4m x 2n), async LDS staging.
#define GBM 64
#define GBN 64
#define GBK 32

template <bool BF16OUT>
__global__ __launch_bounds__(256) void gemm_bf16_kernel(
    const unsigned short* __restrict__ X,    // [M,K] bf16 row-major
    const unsigned short* __restrict__ Wt,   // [N,K] bf16 row-major (pre-transposed)
    const float* __restrict__ bias, void* __restrict__ Y,
    int M, int N, int K, float outScale)
{
    __shared__ unsigned short As[GBM][GBK];   // [m][k]
    __shared__ unsigned short Bs[GBN][GBK];   // [n][k]

    const int tid  = threadIdx.x;
    const int lane = tid & 31;
    const int wave = tid >> 5;
    const int wm   = wave >> 1;
    const int wn   = wave & 1;
    const int half = lane >> 4;
    const int n16  = lane & 15;
    const int bm0  = blockIdx.y * GBM;
    const int bn0  = blockIdx.x * GBN;

    const unsigned asBase = lds_off_of(&As[0][0]);
    const unsigned bsBase = lds_off_of(&Bs[0][0]);
    const int srow = tid >> 2;            // 0..63 (tile row staged by this lane)
    const int scol = (tid & 3) * 8;       // element col of 16B chunk

    v8f acc0 = {}; v8f acc1 = {};

    for (int k0 = 0; k0 < K; k0 += GBK) {
        // async-stage both 4KB tiles: one b128 per lane per tile
        async_copy_b128(asBase + tid * 16, X  + (size_t)(bm0 + srow) * K + k0 + scol);
        async_copy_b128(bsBase + tid * 16, Wt + (size_t)(bn0 + srow) * K + k0 + scol);
        wait_async0();
        __syncthreads();

        // A fragment: rows wm*16+n16; elems: two contiguous 16B runs
        BF16x16 a;
        __builtin_memcpy(&a.s[0], &As[wm * 16 + n16][16 * half],      16);
        __builtin_memcpy(&a.s[8], &As[wm * 16 + n16][16 * half + 8], 16);
        // B fragments: lane n, elems K=16*half+e contiguous (32B run)
        BF16x16 b0, b1;
        __builtin_memcpy(&b0.s[0], &Bs[wn * 32 + n16][16 * half],      32);
        __builtin_memcpy(&b1.s[0], &Bs[wn * 32 + 16 + n16][16 * half], 32);

        acc0 = __builtin_amdgcn_wmma_f32_16x16x32_bf16(false, a.v, false, b0.v,
                                                       (short)0, acc0, false, false);
        acc1 = __builtin_amdgcn_wmma_f32_16x16x32_bf16(false, a.v, false, b1.v,
                                                       (short)0, acc1, false, false);
        __syncthreads();
    }

    const int row0 = bm0 + wm * 16 + 8 * half;
    const int col0 = bn0 + wn * 32 + n16;
    const float bb0 = bias[col0];
    const float bb1 = bias[col0 + 16];
    if (BF16OUT) {
        unsigned short* Yb = (unsigned short*)Y;
        #pragma unroll
        for (int r = 0; r < 8; ++r) {
            Yb[(size_t)(row0 + r) * N + col0]      = f2bfu((acc0[r] + bb0) * outScale);
            Yb[(size_t)(row0 + r) * N + col0 + 16] = f2bfu((acc1[r] + bb1) * outScale);
        }
    } else {
        float* Yf = (float*)Y;
        #pragma unroll
        for (int r = 0; r < 8; ++r) {
            Yf[(size_t)(row0 + r) * N + col0]      = (acc0[r] + bb0) * outScale;
            Yf[(size_t)(row0 + r) * N + col0 + 16] = (acc1[r] + bb1) * outScale;
        }
    }
}

// ---- Causal flash attention (bf16 Q/K/V in, bf16 ctx out) -------------------
// 128 thr = 4 waves; wave w: queries qb+16w..+15. dh=64, key chunks of 32.
// Q pre-scaled by dh^-0.5 in the Q-projection epilogue.
__global__ __launch_bounds__(128) void flash_attn_kernel(
    const unsigned short* __restrict__ Q, const unsigned short* __restrict__ Kp,
    const unsigned short* __restrict__ Vp, unsigned short* __restrict__ O)
{
    const int L = 2048, D = 1024, DH = 64;
    __shared__ unsigned short Ks[32][64];       // [key][d]
    __shared__ unsigned short Vst[64][32];      // [d][key]
    __shared__ unsigned short Ps[4][16][32];    // per-wave P

    const int tid  = threadIdx.x;
    const int lane = tid & 31;
    const int wave = tid >> 5;
    const int half = lane >> 4;
    const int n16  = lane & 15;
    const int qb   = blockIdx.x * 64;
    const int h    = blockIdx.y;
    const int b    = blockIdx.z;

    const unsigned short* Qb = Q  + (size_t)b * L * D + (size_t)h * DH;
    const unsigned short* Kb = Kp + (size_t)b * L * D + (size_t)h * DH;
    const unsigned short* Vb = Vp + (size_t)b * L * D + (size_t)h * DH;
    unsigned short*       Ob = O  + (size_t)b * L * D + (size_t)h * DH;

    const unsigned ksBase = lds_off_of(&Ks[0][0]);

    // Q fragments (A layout): two contiguous 16B runs per 32-deep chunk
    BF16x16 qa[2];
    const int qrow = qb + wave * 16 + n16;
    #pragma unroll
    for (int c = 0; c < 2; ++c) {
        __builtin_memcpy(&qa[c].s[0], Qb + (size_t)qrow * D + c * 32 + 16 * half,      16);
        __builtin_memcpy(&qa[c].s[8], Qb + (size_t)qrow * D + c * 32 + 16 * half + 8, 16);
    }

    v8f acc0 = {}, acc1 = {}, acc2 = {}, acc3 = {};
    float mrow[8], lrow[8];
    #pragma unroll
    for (int r = 0; r < 8; ++r) { mrow[r] = -1e30f; lrow[r] = 0.0f; }

    const int kend = qb + 64;                   // causal upper bound
    for (int kt = 0; kt < kend; kt += 32) {
        // K tile: async copy, 2 x b128 per lane (4KB)
        #pragma unroll
        for (int j = 0; j < 2; ++j) {
            int lin = (tid + j * 128) * 16;     // byte offset in tile
            int kk  = lin >> 7;                 // 128B per key row
            int ce  = (lin & 127) >> 1;         // element col
            async_copy_b128(ksBase + lin, Kb + (size_t)(kt + kk) * D + ce);
        }
        // V tile transposed: vector load + ds scatter (overlaps the async)
        #pragma unroll
        for (int j = 0; j < 2; ++j) {
            int lin = tid * 8 + j * 1024;       // element index in 32x64 tile
            int kk  = lin >> 6;
            int d0  = lin & 63;
            unsigned short tmp[8];
            __builtin_memcpy(tmp, Vb + (size_t)(kt + kk) * D + d0, 16);
            #pragma unroll
            for (int i = 0; i < 8; ++i) Vst[d0 + i][kk] = tmp[i];
        }
        wait_async0();
        __syncthreads();

        // scores: two 16-key sub-tiles, dh=64 -> 2 chunks => 4 WMMAs
        v8f s0 = {}, s1 = {};
        #pragma unroll
        for (int c = 0; c < 2; ++c) {
            BF16x16 bk0, bk1;
            __builtin_memcpy(&bk0.s[0], &Ks[n16][c * 32 + 16 * half],      32);
            __builtin_memcpy(&bk1.s[0], &Ks[16 + n16][c * 32 + 16 * half], 32);
            s0 = __builtin_amdgcn_wmma_f32_16x16x32_bf16(false, qa[c].v, false, bk0.v,
                                                         (short)0, s0, false, false);
            s1 = __builtin_amdgcn_wmma_f32_16x16x32_bf16(false, qa[c].v, false, bk1.v,
                                                         (short)0, s1, false, false);
        }

        // causal mask + online softmax (C layout: row r+8*half, keys n16/n16+16)
        const int qg  = qb + wave * 16 + 8 * half;
        const int kg0 = kt + n16;
        const int kg1 = kt + 16 + n16;
        #pragma unroll
        for (int r = 0; r < 8; ++r) {
            float v0 = (kg0 <= qg + r) ? s0[r] : -1e30f;
            float v1 = (kg1 <= qg + r) ? s1[r] : -1e30f;
            float mx   = rmax16(fmaxf(v0, v1));
            float mnew = fmaxf(mrow[r], mx);
            float sc   = __expf(mrow[r] - mnew);
            float e0   = __expf(v0 - mnew);
            float e1   = __expf(v1 - mnew);
            lrow[r] = lrow[r] * sc + rsum16(e0 + e1);
            mrow[r] = mnew;
            acc0[r] *= sc; acc1[r] *= sc; acc2[r] *= sc; acc3[r] *= sc;
            Ps[wave][r + 8 * half][n16]      = f2bfu(e0);
            Ps[wave][r + 8 * half][16 + n16] = f2bfu(e1);
        }
        __syncthreads();

        // P (A layout) x V (B layout from Vst): 4 WMMAs over d tiles
        BF16x16 pa;
        __builtin_memcpy(&pa.s[0], &Ps[wave][n16][16 * half],      16);
        __builtin_memcpy(&pa.s[8], &Ps[wave][n16][16 * half + 8], 16);
        {
            BF16x16 bv0, bv1, bv2, bv3;
            __builtin_memcpy(&bv0.s[0], &Vst[n16][16 * half],      32);
            __builtin_memcpy(&bv1.s[0], &Vst[16 + n16][16 * half], 32);
            __builtin_memcpy(&bv2.s[0], &Vst[32 + n16][16 * half], 32);
            __builtin_memcpy(&bv3.s[0], &Vst[48 + n16][16 * half], 32);
            acc0 = __builtin_amdgcn_wmma_f32_16x16x32_bf16(false, pa.v, false, bv0.v,
                                                           (short)0, acc0, false, false);
            acc1 = __builtin_amdgcn_wmma_f32_16x16x32_bf16(false, pa.v, false, bv1.v,
                                                           (short)0, acc1, false, false);
            acc2 = __builtin_amdgcn_wmma_f32_16x16x32_bf16(false, pa.v, false, bv2.v,
                                                           (short)0, acc2, false, false);
            acc3 = __builtin_amdgcn_wmma_f32_16x16x32_bf16(false, pa.v, false, bv3.v,
                                                           (short)0, acc3, false, false);
        }
        __syncthreads();   // before next chunk overwrites Ks/Vst
    }

    // normalize and store bf16 context
    const int orow = qb + wave * 16 + 8 * half;
    #pragma unroll
    for (int r = 0; r < 8; ++r) {
        float inv = 1.0f / lrow[r];
        Ob[(size_t)(orow + r) * D + n16]      = f2bfu(acc0[r] * inv);
        Ob[(size_t)(orow + r) * D + n16 + 16] = f2bfu(acc1[r] * inv);
        Ob[(size_t)(orow + r) * D + n16 + 32] = f2bfu(acc2[r] * inv);
        Ob[(size_t)(orow + r) * D + n16 + 48] = f2bfu(acc3[r] * inv);
    }
}

// ---- launch -----------------------------------------------------------------
extern "C" void kernel_launch(void* const* d_in, const int* in_sizes, int n_in,
                              void* d_out, int out_size, void* d_ws, size_t ws_size,
                              hipStream_t stream) {
    const float* query = (const float*)d_in[0];
    const float* key   = (const float*)d_in[1];
    const float* value = (const float*)d_in[2];
    const float* Wq = (const float*)d_in[3];  const float* bq = (const float*)d_in[4];
    const float* Wk = (const float*)d_in[5];  const float* bk = (const float*)d_in[6];
    const float* Wv = (const float*)d_in[7];  const float* bv = (const float*)d_in[8];
    const float* Wo = (const float*)d_in[9];  const float* bo = (const float*)d_in[10];
    float* out = (float*)d_out;

    const int B = 2, L = 2048, Dm = 1024, H = 16;
    const int M = B * L;                          // 4096
    const size_t plane = (size_t)M * Dm;          // 4M elems
    const size_t wsz   = (size_t)Dm * Dm;         // 1M elems
    const float qscale = 0.125f;                  // dh^-0.5

    // bf16 workspace layout (ushort units), total 64 MB
    unsigned short* xq = (unsigned short*)d_ws;
    unsigned short* xk = xq + plane;
    unsigned short* xv = xk + plane;
    unsigned short* wqT = xv + plane;
    unsigned short* wkT = wqT + wsz;
    unsigned short* wvT = wkT + wsz;
    unsigned short* woT = wvT + wsz;
    unsigned short* qbuf = woT + wsz;
    unsigned short* kbuf = qbuf + plane;
    unsigned short* vbuf = kbuf + plane;
    unsigned short* cbuf = vbuf + plane;

    // 1) one-time conversions
    {
        int n4 = (int)(plane / 4);
        dim3 g((n4 + 255) / 256), blk(256);
        cvt_bf16_kernel<<<g, blk, 0, stream>>>((const float4*)query, (uint2*)xq, n4);
        cvt_bf16_kernel<<<g, blk, 0, stream>>>((const float4*)key,   (uint2*)xk, n4);
        cvt_bf16_kernel<<<g, blk, 0, stream>>>((const float4*)value, (uint2*)xv, n4);
    }
    {
        dim3 g(Dm / 64, Dm / 64), blk(256);
        cvt_transpose_kernel<<<g, blk, 0, stream>>>(Wq, wqT, Dm, Dm);
        cvt_transpose_kernel<<<g, blk, 0, stream>>>(Wk, wkT, Dm, Dm);
        cvt_transpose_kernel<<<g, blk, 0, stream>>>(Wv, wvT, Dm, Dm);
        cvt_transpose_kernel<<<g, blk, 0, stream>>>(Wo, woT, Dm, Dm);
    }

    // 2) projections (Q pre-scaled by dh^-0.5), bf16 outputs
    dim3 gemmGrid(Dm / GBN, M / GBM), gemmBlock(256);
    gemm_bf16_kernel<true><<<gemmGrid, gemmBlock, 0, stream>>>(xq, wqT, bq, qbuf, M, Dm, Dm, qscale);
    gemm_bf16_kernel<true><<<gemmGrid, gemmBlock, 0, stream>>>(xk, wkT, bk, kbuf, M, Dm, Dm, 1.0f);
    gemm_bf16_kernel<true><<<gemmGrid, gemmBlock, 0, stream>>>(xv, wvT, bv, vbuf, M, Dm, Dm, 1.0f);

    // 3) causal flash attention -> bf16 context
    dim3 attnGrid(L / 64, H, B);
    flash_attn_kernel<<<attnGrid, 128, 0, stream>>>(qbuf, kbuf, vbuf, cbuf);

    // 4) output projection, fp32 result
    gemm_bf16_kernel<false><<<gemmGrid, gemmBlock, 0, stream>>>(cbuf, woT, bo, out, M, Dm, Dm, 1.0f);
}
